// ScaledDotProductAttention_15135464751719
// MI455X (gfx1250) — matrix operand order repaired and checked by reading, not compiled, not run on hardware
//
#include <hip/hip_runtime.h>
#include <hip/hip_bf16.h>

// ---------------------------------------------------------------------------
// Problem constants (from reference):
//   B=32, Tq=Tk=1024, d_q=512, d_k=256, fg=128, scale = 1/sqrt(256) = 1/16
// Output tuple: ctx [B,Tq,256] f32  followed by  attn [B,Tq,Tk] f32
// ---------------------------------------------------------------------------
#define BB   32
#define TQ   1024
#define TK   1024
#define DQ   512
#define DKV  256
#define FG   128
#define SCALE 0.0625f
#define LOG2E 1.44269504088896340736f

typedef __attribute__((ext_vector_type(16))) __bf16 v16bf;
typedef __attribute__((ext_vector_type(8)))  __bf16 v8bf;
typedef __attribute__((ext_vector_type(8)))  float  v8f;
typedef __attribute__((ext_vector_type(4)))  unsigned int v4u;
typedef __attribute__((ext_vector_type(8)))  int v8i;
typedef __attribute__((ext_vector_type(4)))  int v4i;

// CDNA5 16-bit A-matrix 16x32 lane layout (ISA 7.12.2):
//   lanes 0-15  : row = lane,    halves 0-7 -> K=0..7,  halves 8-15 -> K=16..23
//   lanes 16-31 : row = lane-16, halves 0-7 -> K=8..15, halves 8-15 -> K=24..31
// => this code uses contiguous K instead: per lane one 16-half (32 B) run at K = koffA,
//    koffA = (lane&16) ? 16 : 0.  B 32x16, same K order: lane col = lane&15, halves
//    i -> K = i + ((lane&16)?16:0): ONE contiguous 16-half (32 B) run.
// C/D 16x16 f32: VGPR r -> row = r + ((lane&16)?8:0); col = lane&15.

union frag16 { v16bf v; v8bf h[2]; };

#if __has_builtin(__builtin_amdgcn_tensor_load_to_lds) && \
    __has_builtin(__builtin_amdgcn_s_wait_tensorcnt)
#define USE_TDM 1
#else
#define USE_TDM 0
#endif

// ---------------------------------------------------------------------------
// Kernel T1: W [d][128] f32  ->  Wt [128][d] bf16   (tiny one-off)
// ---------------------------------------------------------------------------
__global__ void transpose_w_kernel(const float* __restrict__ W,
                                   __bf16* __restrict__ Wt, int d) {
    int idx = blockIdx.x * 256 + threadIdx.x;
    if (idx >= d * FG) return;
    int r = idx / FG, c = idx - r * FG;      // coalesced read over c
    Wt[c * d + r] = (__bf16)W[idx];
}

// ---------------------------------------------------------------------------
// Kernel T2: v [b][s][d] f32 -> vt [b][d][s] bf16  (LDS-tiled 64x64 transpose)
// Puts V in B-operand-major order: ctx B-frags become contiguous 32 B loads.
// ---------------------------------------------------------------------------
__global__ __launch_bounds__(256) void transpose_v_kernel(
    const float* __restrict__ v, __bf16* __restrict__ vt) {
    __shared__ __bf16 tile[64][66];          // 66-half row pitch: conflict-free
    const int b  = blockIdx.z;
    const int s0 = blockIdx.x * 64;
    const int d0 = blockIdx.y * 64;
    const int j  = threadIdx.x & 63;
    const int i0 = threadIdx.x >> 6;         // 0..3
    #pragma unroll
    for (int k = 0; k < 16; ++k) {
        int i = i0 + k * 4;
        tile[i][j] = (__bf16)v[((long)b * TK + s0 + i) * DKV + d0 + j];
    }
    __syncthreads();
    #pragma unroll
    for (int k = 0; k < 16; ++k) {
        int i = i0 + k * 4;                  // output row = feature d
        vt[((long)b * DKV + d0 + i) * TK + s0 + j] = tile[j][i];
    }
}

// ---------------------------------------------------------------------------
// Kernel 1: projection GEMM  Y[R,128] = X[R,d] @ W[d,128] with W pre-
// transposed to Wt[128][d] bf16. One block = 16 output rows; 8 waves, wave w
// owns one 16-col N-tile. Inner loop: 2x ds_load_b128 (A) + 2x
// global_load_b128 (B) + 1x v_wmma.
// ---------------------------------------------------------------------------
__global__ __launch_bounds__(256) void proj_bf16_kernel(
    const float* __restrict__ X, const __bf16* __restrict__ Wt,
    __bf16* __restrict__ Y, int d)
{
    __shared__ __bf16 As[16][DQ + 8];        // 1040 B row pitch (16 B aligned)

    const int tid  = threadIdx.x;
    const int lane = tid & 31;
    const int w    = tid >> 5;
    const long rowBase = (long)blockIdx.x * 16;

    for (int idx = tid; idx < 16 * d; idx += 256) {
        int r = idx / d, c = idx - r * d;
        As[r][c] = (__bf16)X[(rowBase + r) * d + c];
    }
    __syncthreads();

    const int col   = w * 16 + (lane & 15);
    const int koffA = (lane & 16) ? 16 : 0;
    const int koffB = (lane & 16) ? 16 : 0;
    const int rsel  = (lane & 16) ? 8 : 0;

    const __bf16* arow = &As[lane & 15][0];
    const __bf16* wcol = Wt + (size_t)col * d;

    v8f acc = {};
    for (int k0 = 0; k0 < d; k0 += 32) {
        frag16 a, b;
        a.h[0] = *(const v8bf*)(arow + k0 + koffA);
        a.h[1] = *(const v8bf*)(arow + k0 + koffA + 8);
        b.h[0] = *(const v8bf*)(wcol + k0 + koffB);
        b.h[1] = *(const v8bf*)(wcol + k0 + koffB + 8);
        acc = __builtin_amdgcn_wmma_f32_16x16x32_bf16(
                  false, a.v, false, b.v, (short)0, acc, false, false);
    }
    #pragma unroll
    for (int r = 0; r < 8; ++r)
        Y[(rowBase + r + rsel) * FG + col] = (__bf16)acc[r];
}

// ---------------------------------------------------------------------------
// Kernel 2: fused  scores -> mask -> softmax -> attn write -> ctx = P @ V.
// Block = (batch b, 16 query rows), 4 waves. The qp A-tile (16x128 bf16,
// 4 KB) is staged into LDS by the Tensor Data Mover; each wave owns 16 of
// the 64 score tiles (full 16x1024 fp32 score strip resident in VGPRs).
// ---------------------------------------------------------------------------
__global__ __launch_bounds__(128) void attn_kernel(
    const __bf16* __restrict__ qp,   // [B,TQ,FG]
    const __bf16* __restrict__ kp,   // [B,TK,FG]
    const __bf16* __restrict__ vt,   // [B,DKV,TK]  (transposed v, bf16)
    const int*    __restrict__ mask, // [B,TQ,TK]
    float* __restrict__ ctx,         // [B,TQ,DKV]
    float* __restrict__ attn)        // [B,TQ,TK]
{
    __shared__ __bf16 qTile[16][FG];        // 4 KB, TDM destination
    __shared__ __bf16 pLds[16][TK + 8];     // normalized probs, 16 B-aligned pitch
    __shared__ float  redmax[4][16];
    __shared__ float  redsum[4][16];

    const int tid  = threadIdx.x;
    const int lane = tid & 31;
    const int w    = tid >> 5;
    const int b    = blockIdx.y;
    const int t0   = blockIdx.x * 16;

    const int mrow  = lane & 15;
    const int rsel  = (lane & 16) ? 8 : 0;
    const int koffA = (lane & 16) ? 16 : 0;
    const int koffB = (lane & 16) ? 16 : 0;

    // --- Stage qp rows t0..t0+15 into LDS via TDM (one wave issues) --------
#if USE_TDM
    if (w == 0) {
        unsigned lds_addr = (unsigned)(uintptr_t)&qTile[0][0];
        unsigned long long ga =
            (unsigned long long)(uintptr_t)(qp + (size_t)(b * TQ + t0) * FG);
        // D# group0: count=1 | lds_addr | global_addr | type=2 (ISA 8.3)
        v4u g0 = { 1u, lds_addr, (unsigned)ga,
                   ((unsigned)(ga >> 32) & 0x01FFFFFFu) | (2u << 30) };
        // D# group1: data_size=2B; tensor_dim0=128; tensor_dim1=32768;
        //            tile_dim0=128; tile_dim1=16; tensor_dim0_stride=128 (ISA 8.4)
        v8i g1 = { 0x10000, (int)(128u << 16), (int)0x80000000u,
                   (int)(128u << 16), 16, 128, 0, 0 };
        v4i gz = { 0, 0, 0, 0 };
#if __clang_major__ >= 23
        v8i gz8 = { 0, 0, 0, 0, 0, 0, 0, 0 };
        __builtin_amdgcn_tensor_load_to_lds(g0, g1, gz, gz, gz8, 0);
#else
        __builtin_amdgcn_tensor_load_to_lds(g0, g1, gz, gz, 0);
#endif
        __builtin_amdgcn_s_wait_tensorcnt(0);
    }
#else
    {   // cooperative fallback: 4 KB as 256 x b128
        const uint4* src = (const uint4*)(qp + (size_t)(b * TQ + t0) * FG);
        uint4* dst = (uint4*)&qTile[0][0];
        dst[tid] = src[tid];
        dst[tid + 128] = src[tid + 128];
    }
#endif
    __syncthreads();

    // --- A-fragments of qp from LDS (two b128 reads per kstep) -------------
    v16bf aq[4];
    {
        const __bf16* qr = &qTile[mrow][0];
        #pragma unroll
        for (int ks = 0; ks < 4; ++ks) {
            frag16 f;
            f.h[0] = *(const v8bf*)(qr + ks * 32 + koffA);
            f.h[1] = *(const v8bf*)(qr + ks * 32 + koffA + 8);
            aq[ks] = f.v;
        }
    }

    // --- Scores: 16 tiles per wave, fully resident in VGPRs ----------------
    v8f sc[16];
    const long mbase = ((long)b * TQ + t0) * TK;
    #pragma unroll
    for (int tIdx = 0; tIdx < 16; ++tIdx) {
        const int s0 = (tIdx * 4 + w) * 16;
        const __bf16* krow = kp + ((long)(b * TK + s0 + mrow)) * FG;
        if (tIdx < 15)  // pull next kp strip (global_prefetch_b8)
            __builtin_prefetch(kp + ((long)(b * TK + s0 + 64 + mrow)) * FG, 0, 3);
        v8f c = {};
        #pragma unroll
        for (int ks = 0; ks < 4; ++ks) {
            frag16 bk;  // B-frag of kp^T: one contiguous 32 B run per lane
            const v8bf* p = (const v8bf*)(krow + ks * 32 + koffB);
            bk.h[0] = p[0];
            bk.h[1] = p[1];
            c = __builtin_amdgcn_wmma_f32_16x16x32_bf16(
                    false, aq[ks], false, bk.v, (short)0, c, false, false);
        }
        const int colS = s0 + mrow;
        #pragma unroll
        for (int r = 0; r < 8; ++r) {
            const int m = mask[mbase + (long)(r + rsel) * TK + colS];
            c[r] = m ? (c[r] * SCALE) : -1.0e9f;
        }
        sc[tIdx] = c;
    }

    // --- Row max: lane-local over tiles, shfl over 16-lane group, LDS over waves
    float pm[8];
    #pragma unroll
    for (int r = 0; r < 8; ++r) {
        float m = sc[0][r];
        #pragma unroll
        for (int t = 1; t < 16; ++t) m = fmaxf(m, sc[t][r]);
        pm[r] = m;
    }
    #pragma unroll
    for (int off = 1; off < 16; off <<= 1)
        #pragma unroll
        for (int r = 0; r < 8; ++r)
            pm[r] = fmaxf(pm[r], __shfl_xor(pm[r], off, 32));
    if ((lane & 15) == 0)
        #pragma unroll
        for (int r = 0; r < 8; ++r) redmax[w][r + rsel] = pm[r];
    __syncthreads();
    float rowmax[8];
    #pragma unroll
    for (int r = 0; r < 8; ++r) {
        float m = redmax[0][r + rsel];
        #pragma unroll
        for (int ww = 1; ww < 4; ++ww) m = fmaxf(m, redmax[ww][r + rsel]);
        rowmax[r] = m;
    }

    // --- exp (v_exp_f32) + row sums ----------------------------------------
    float ps[8];
    #pragma unroll
    for (int r = 0; r < 8; ++r) ps[r] = 0.0f;
    #pragma unroll
    for (int tIdx = 0; tIdx < 16; ++tIdx)
        #pragma unroll
        for (int r = 0; r < 8; ++r) {
            float p = __builtin_exp2f((sc[tIdx][r] - rowmax[r]) * LOG2E);
            sc[tIdx][r] = p;
            ps[r] += p;
        }
    #pragma unroll
    for (int off = 1; off < 16; off <<= 1)
        #pragma unroll
        for (int r = 0; r < 8; ++r)
            ps[r] += __shfl_xor(ps[r], off, 32);
    if ((lane & 15) == 0)
        #pragma unroll
        for (int r = 0; r < 8; ++r) redsum[w][r + rsel] = ps[r];
    __syncthreads();
    float rinv[8];
    #pragma unroll
    for (int r = 0; r < 8; ++r) {
        float s = redsum[0][r + rsel];
        #pragma unroll
        for (int ww = 1; ww < 4; ++ww) s += redsum[ww][r + rsel];
        rinv[r] = 1.0f / s;
    }

    // --- Write attn (fp32 -> HBM) and bf16 probs into LDS ------------------
    #pragma unroll
    for (int tIdx = 0; tIdx < 16; ++tIdx) {
        const int colS = (tIdx * 4 + w) * 16 + mrow;
        #pragma unroll
        for (int r = 0; r < 8; ++r) {
            float p = sc[tIdx][r] * rinv[r];
            attn[mbase + (long)(r + rsel) * TK + colS] = p;
            pLds[r + rsel][colS] = (__bf16)p;
        }
    }
    __syncthreads();

    // --- ctx = P @ V: wave w owns N-tiles 4w..4w+3; K = 1024 (32 WMMA steps)
    v8f cacc[4];
    {
        v8f z = {};
        #pragma unroll
        for (int j = 0; j < 4; ++j) cacc[j] = z;
    }
    const __bf16* prow = &pLds[mrow][0];
    #pragma unroll 1
    for (int ks = 0; ks < 32; ++ks) {
        frag16 pa;  // A-frag of probabilities: two b128 LDS reads
        pa.h[0] = *(const v8bf*)(prow + ks * 32 + koffA);
        pa.h[1] = *(const v8bf*)(prow + ks * 32 + koffA + 8);
        #pragma unroll
        for (int j = 0; j < 4; ++j) {
            const int colV = (w * 4 + j) * 16 + mrow;
            const __bf16* vrow = vt + ((long)b * DKV + colV) * TK + ks * 32 + koffB;
            frag16 bv;  // B-frag of v^T: one contiguous 32 B run per lane
            bv.h[0] = *(const v8bf*)(vrow);
            bv.h[1] = *(const v8bf*)(vrow + 8);
            cacc[j] = __builtin_amdgcn_wmma_f32_16x16x32_bf16(
                          false, pa.v, false, bv.v, (short)0, cacc[j], false, false);
        }
    }
    #pragma unroll
    for (int j = 0; j < 4; ++j) {
        const int colV = (w * 4 + j) * 16 + mrow;
        #pragma unroll
        for (int r = 0; r < 8; ++r)
            ctx[((long)(b * TQ) + t0 + r + rsel) * DKV + colV] = cacc[j][r];
    }
}

// ---------------------------------------------------------------------------
// Launch
// ---------------------------------------------------------------------------
extern "C" void kernel_launch(void* const* d_in, const int* in_sizes, int n_in,
                              void* d_out, int out_size, void* d_ws, size_t ws_size,
                              hipStream_t stream) {
    const float* q    = (const float*)d_in[0];
    const float* k    = (const float*)d_in[1];
    const float* v    = (const float*)d_in[2];
    const int*   mask = (const int*)d_in[3];
    const float* Wq   = (const float*)d_in[4];
    const float* Wk   = (const float*)d_in[5];

    float* ctx  = (float*)d_out;                      // [B,TQ,DKV]
    float* attn = ctx + (long)BB * TQ * DKV;          // [B,TQ,TK]

    // Workspace: qp (8MB) | kp (8MB) | vt (16MB) | Wqt (128KB) | Wkt (64KB)
    char*   ws  = (char*)d_ws;
    size_t  off = 0;
    __bf16* qp  = (__bf16*)(ws + off); off += (size_t)BB * TQ * FG * 2;
    __bf16* kp  = (__bf16*)(ws + off); off += (size_t)BB * TK * FG * 2;
    __bf16* vt  = (__bf16*)(ws + off); off += (size_t)BB * DKV * TK * 2;
    __bf16* Wqt = (__bf16*)(ws + off); off += (size_t)DQ * FG * 2;
    __bf16* Wkt = (__bf16*)(ws + off);

    transpose_w_kernel<<<(DQ * FG + 255) / 256, 256, 0, stream>>>(Wq, Wqt, DQ);
    transpose_w_kernel<<<(DKV * FG + 255) / 256, 256, 0, stream>>>(Wk, Wkt, DKV);
    transpose_v_kernel<<<dim3(TK / 64, DKV / 64, BB), 256, 0, stream>>>(v, vt);

    proj_bf16_kernel<<<(BB * TQ) / 16, 256, 0, stream>>>(q, Wqt, qp, DQ);
    proj_bf16_kernel<<<(BB * TK) / 16, 256, 0, stream>>>(k, Wkt, kp, DKV);

    dim3 g(TQ / 16, BB);
    attn_kernel<<<g, 128, 0, stream>>>(qp, kp, vt, mask, ctx, attn);
}
